// SharedMemoryAttention_27986006901169
// MI455X (gfx1250) — compile-verified, matrix operand order checked by reading, not run on hardware
//
#include <hip/hip_runtime.h>
#include <hip/hip_bf16.h>

#define B_ 32
#define E_ 512
#define F_ 4096
#define M_ 128
#define S_ 256

typedef __bf16 v16bf __attribute__((ext_vector_type(16)));
typedef float  v8f   __attribute__((ext_vector_type(8)));

union Frag {
    v16bf v;
    uint4 q[2];
    unsigned short u[16];
};

__device__ __forceinline__ unsigned short f2bf(float f) {
    unsigned int u = __float_as_uint(f);
    u += 0x7fffu + ((u >> 16) & 1u);          // round-to-nearest-even
    return (unsigned short)(u >> 16);
}

// gfx1250 async global->LDS path (upstream clang builtins); guarded so the
// kernel still compiles (with synchronous staging) if the toolchain lacks them.
#if defined(__AMDGCN__) && __has_builtin(__builtin_amdgcn_global_load_async_to_lds_b32) && \
    __has_builtin(__builtin_amdgcn_s_wait_asynccnt)
#define HAS_ASYNC_LDS 1
#else
#define HAS_ASYNC_LDS 0
#endif

// ---------------------------------------------------------------------------
// Prep: convert the small, heavily reused parameter matrices to bf16 once.
//   ws[0      .. 65535 ]  = W_q   bf16 [128 m][512 e]   (row-major, B-frag friendly)
//   ws[65536  .. 98303 ]  = keys  bf16 [256 s][128 m]   (row-major, B-frag friendly)
//   ws[98304  .. 131071]  = V^T   bf16 [128 m][256 s]   (transposed for B-frags)
// ---------------------------------------------------------------------------
__global__ void smem_attn_prep(const float* __restrict__ Wq,
                               const float* __restrict__ keys,
                               const float* __restrict__ values,
                               unsigned short* __restrict__ ws) {
    int i = blockIdx.x * 256 + threadIdx.x;   // 0 .. 65535
    ws[i] = f2bf(Wq[i]);
    if (i < S_ * M_) {
        ws[65536 + i] = f2bf(keys[i]);
        int m = i >> 8, s = i & 255;
        ws[98304 + m * S_ + s] = f2bf(values[(size_t)s * M_ + m]);
    }
}

// ---------------------------------------------------------------------------
// Main kernel: one block = (batch b, 128-column f-tile); 8 waves, each wave
// owns 16 f-rows and runs the full Q -> scores -> softmax -> retrieve pipeline
// with bf16 WMMA (f32 accumulate).
// ---------------------------------------------------------------------------
__global__ __launch_bounds__(256)
void smem_attn_main(const float* __restrict__ x,
                    const float* __restrict__ bq,
                    const unsigned short* __restrict__ wqb,   // [128][512] bf16
                    const unsigned short* __restrict__ keyb,  // [256][128] bf16
                    const unsigned short* __restrict__ valb,  // [128][256] bf16 (V^T)
                    float* __restrict__ out) {
    // 64 KB LDS, regions alias across pipeline phases (barrier-separated):
    //   QS: [8 waves][16 f][128 m] bf16  bytes 0     .. 32767
    //   XS: [128 f][32 e]          f32   bytes 32768 .. 49151
    //   PS: [8 waves][16 f][256 s] bf16  bytes 0     .. 65535 (after barrier)
    __shared__ alignas(16) unsigned short sm[32768];
    const int QS = 0, PS = 0;
    float* xsf = (float*)(sm + 16384);        // byte offset 32768

    const int tid  = threadIdx.x;
    const int wv   = tid >> 5;
    const int lane = tid & 31;
    const int l16  = lane & 15;
    const int half = lane >> 4;

    const int b     = blockIdx.x >> 5;
    const int fbase = (blockIdx.x & 31) * 128;

    // ---------------- Phase 1: Q = X^T * Wq^T  (K = 512 in 16 steps) --------
    v8f qacc[8];
#pragma unroll
    for (int t = 0; t < 8; ++t) qacc[t] = {};

    const int sf  = tid & 127;         // staged f column
    const int er0 = (tid >> 7) * 16;   // staged e row group

    for (int eb = 0; eb < E_; eb += 32) {
        __syncthreads();               // previous tile fully consumed
#pragma unroll
        for (int ii = 0; ii < 16; ++ii) {
            int e = er0 + ii;
            const float* gp = &x[((size_t)b * E_ + (eb + e)) * F_ + fbase + sf];
#if HAS_ASYNC_LDS
            __builtin_amdgcn_global_load_async_to_lds_b32(
                (int*)gp, (int*)&xsf[sf * 32 + e], 0, 0);
#else
            xsf[sf * 32 + e] = *gp;    // xs[f][e], 128 B rows
#endif
        }
#if HAS_ASYNC_LDS
        __builtin_amdgcn_s_wait_asynccnt(0);
#endif
        __syncthreads();

        // A frag: 16 f-rows x 32 e, f32 in LDS -> bf16 at fragment build
        Frag a;
        {
            const float* row = xsf + (wv * 16 + l16) * 32 + half * 8;
            float4 c0 = *(const float4*)(row);
            float4 c1 = *(const float4*)(row + 4);
            float4 c2 = *(const float4*)(row + 16);
            float4 c3 = *(const float4*)(row + 20);
            a.u[0]  = f2bf(c0.x); a.u[1]  = f2bf(c0.y);
            a.u[2]  = f2bf(c0.z); a.u[3]  = f2bf(c0.w);
            a.u[4]  = f2bf(c1.x); a.u[5]  = f2bf(c1.y);
            a.u[6]  = f2bf(c1.z); a.u[7]  = f2bf(c1.w);
            a.u[8]  = f2bf(c2.x); a.u[9]  = f2bf(c2.y);
            a.u[10] = f2bf(c2.z); a.u[11] = f2bf(c2.w);
            a.u[12] = f2bf(c3.x); a.u[13] = f2bf(c3.y);
            a.u[14] = f2bf(c3.z); a.u[15] = f2bf(c3.w);
        }
#pragma unroll
        for (int t = 0; t < 8; ++t) {  // 8 m-tiles of 16
            Frag bf;
            const unsigned short* p = wqb + (size_t)(t * 16 + l16) * 512 + eb + half * 8;
            bf.q[0] = *(const uint4*)p;
            bf.q[1] = *(const uint4*)(p + 16);
            qacc[t] = __builtin_amdgcn_wmma_f32_16x16x32_bf16(
                false, a.v, false, bf.v, (short)0, qacc[t], false, false);
        }
    }

    // bias add + spill Q to LDS as bf16 in A-fragment layout qs[f][m]
#pragma unroll
    for (int t = 0; t < 8; ++t) {
        float bias = bq[t * 16 + l16];
#pragma unroll
        for (int r = 0; r < 8; ++r) {
            float qv = qacc[t][r] + bias;
            sm[QS + wv * 2048 + (r + 8 * half) * 128 + t * 16 + l16] = f2bf(qv);
        }
    }

    // ---------------- Phase 2: S = Q * K^T  (16 slot-tiles, K = 128) --------
    v8f sc[16];
#pragma unroll
    for (int n = 0; n < 16; ++n) {
        v8f acc = {};
#pragma unroll
        for (int kk = 0; kk < 4; ++kk) {
            Frag a, bf;
            const int ab = QS + wv * 2048 + l16 * 128 + kk * 32 + half * 8;
            a.q[0] = *(const uint4*)&sm[ab];
            a.q[1] = *(const uint4*)&sm[ab + 16];
            const unsigned short* p = keyb + (size_t)(n * 16 + l16) * 128 + kk * 32 + half * 8;
            bf.q[0] = *(const uint4*)p;
            bf.q[1] = *(const uint4*)(p + 16);
            acc = __builtin_amdgcn_wmma_f32_16x16x32_bf16(
                false, a.v, false, bf.v, (short)0, acc, false, false);
        }
        sc[n] = acc;
    }

    // Row softmax over 256 slots. C-tile layout: lane%16 = slot within tile,
    // VGPR r = f-row (r for lanes 0-15, r+8 for lanes 16-31), so xor masks
    // 1/2/4/8 reduce exactly over one matrix row.
#pragma unroll
    for (int r = 0; r < 8; ++r) {
        float mx = -3.402823466e38f;
#pragma unroll
        for (int n = 0; n < 16; ++n) mx = fmaxf(mx, sc[n][r]);
        mx = fmaxf(mx, __shfl_xor(mx, 1, 32));
        mx = fmaxf(mx, __shfl_xor(mx, 2, 32));
        mx = fmaxf(mx, __shfl_xor(mx, 4, 32));
        mx = fmaxf(mx, __shfl_xor(mx, 8, 32));
        float su = 0.f;
#pragma unroll
        for (int n = 0; n < 16; ++n) {
            float p = __expf(sc[n][r] - mx);
            sc[n][r] = p;
            su += p;
        }
        su += __shfl_xor(su, 1, 32);
        su += __shfl_xor(su, 2, 32);
        su += __shfl_xor(su, 4, 32);
        su += __shfl_xor(su, 8, 32);
        float inv = 1.0f / su;
#pragma unroll
        for (int n = 0; n < 16; ++n) sc[n][r] *= inv;
    }

    __syncthreads();   // all waves done reading QS/XS before PS overwrite

    // P -> LDS (bf16, A-fragment layout ps[f][s])
#pragma unroll
    for (int n = 0; n < 16; ++n)
#pragma unroll
        for (int r = 0; r < 8; ++r)
            sm[PS + wv * 4096 + (r + 8 * half) * 256 + n * 16 + l16] = f2bf(sc[n][r]);

    // ---------------- Phase 3: R = P * V  (8 m-tiles, K = 256) --------------
#pragma unroll
    for (int t = 0; t < 8; ++t) {
        v8f acc = {};
#pragma unroll
        for (int kk = 0; kk < 8; ++kk) {
            Frag a, bf;
            const int ab = PS + wv * 4096 + l16 * 256 + kk * 32 + half * 8;
            a.q[0] = *(const uint4*)&sm[ab];
            a.q[1] = *(const uint4*)&sm[ab + 16];
            const unsigned short* p = valb + (size_t)(t * 16 + l16) * 256 + kk * 32 + half * 8;
            bf.q[0] = *(const uint4*)p;
            bf.q[1] = *(const uint4*)(p + 16);
            acc = __builtin_amdgcn_wmma_f32_16x16x32_bf16(
                false, a.v, false, bf.v, (short)0, acc, false, false);
        }
        // out[b][m][f]: lane column = m, VGPR r = f-row -> 8 contiguous floats
        size_t ob = ((size_t)b * M_ + t * 16 + l16) * F_ + fbase + wv * 16 + 8 * half;
#pragma unroll
        for (int r = 0; r < 8; ++r) out[ob + r] = acc[r];
    }
}

extern "C" void kernel_launch(void* const* d_in, const int* in_sizes, int n_in,
                              void* d_out, int out_size, void* d_ws, size_t ws_size,
                              hipStream_t stream) {
    const float* x      = (const float*)d_in[0];
    const float* Wq     = (const float*)d_in[1];
    const float* bqv    = (const float*)d_in[2];
    const float* keys   = (const float*)d_in[3];
    const float* values = (const float*)d_in[4];
    unsigned short* ws  = (unsigned short*)d_ws;

    smem_attn_prep<<<256, 256, 0, stream>>>(Wq, keys, values, ws);
    smem_attn_main<<<dim3(B_ * (F_ / 128)), dim3(256), 0, stream>>>(
        x, bqv, ws, ws + 65536, ws + 98304, (float*)d_out);
}
